// TGCNCell_70824010711821
// MI455X (gfx1250) — matrix operand order repaired and checked by reading, not compiled, not run on hardware
//
#include <hip/hip_runtime.h>
#include <hip/hip_bf16.h>

// ---------------------------------------------------------------------------
// T-GCN cell for MI455X (gfx1250, wave32, WMMA bf16 path)
//   z  = relu(GCNConv(x))            (GEMM + deg-norm gather/scatter)
//   r  = sigmoid([z,h] @ W_r + b_r)
//   u  = sigmoid([z,h] @ W_u + b_u)
//   c  = tanh([z, r*h] @ W_c + b_c)
//   h' = u*h + (1-u)*c
// ---------------------------------------------------------------------------

typedef __attribute__((ext_vector_type(16))) __bf16 v16bf;
typedef __attribute__((ext_vector_type(8)))  float  v8f;

#define FDIM 128   // F_IN == G == H
#define KCAT 256   // G + H

union FragU { v16bf v; uint4 q[2]; };

// A-fragment: 16x32 bf16, row-major source with rowStride elements.
// Lane t: M = t&15, K-halves at kb + 8*(t>>4) and kb + 16 + 8*(t>>4).
__device__ __forceinline__ v16bf load_a_frag(const __bf16* rowBase, int rowStride,
                                             int kb, int lane) {
  const int m = lane & 15, half = lane >> 4;
  const __bf16* p = rowBase + (size_t)m * rowStride + kb + 8 * half;
  FragU f;
  f.q[0] = *(const uint4*)(p);
  f.q[1] = *(const uint4*)(p + 16);
  return f.v;
}

// B-fragment: 32x16 bf16 from transposed weights Wt[N][K] (K contiguous).
// Lane t: N = col0 + (t&15), K-halves at kb + 8*(t>>4) and kb + 16 + 8*(t>>4).
__device__ __forceinline__ v16bf load_b_frag(const __bf16* wt, int Kdim,
                                             int col0, int kb, int lane) {
  const int n = lane & 15, half = lane >> 4;
  const __bf16* p = wt + (size_t)(col0 + n) * Kdim + kb + 8 * half;
  FragU f;
  f.q[0] = *(const uint4*)(p);
  f.q[1] = *(const uint4*)(p + 16);
  return f.v;
}

__device__ __forceinline__ float sigmoidf_(float x) {
  return 1.0f / (1.0f + __expf(-x));
}

// --------------------------- small prep kernels ----------------------------

// W [K][Ncols] f32 -> Wt [Ncols][K] bf16 (transposed for B-fragment loads)
__global__ void k_cvt_weight(const float* __restrict__ w, __bf16* __restrict__ wt,
                             int K, int Ncols) {
  int idx = blockIdx.x * blockDim.x + threadIdx.x;
  if (idx >= K * Ncols) return;
  int k = idx / Ncols, n = idx % Ncols;
  wt[(size_t)n * K + k] = (__bf16)w[idx];
}

__global__ void k_cast_x(const float* __restrict__ x, __bf16* __restrict__ xb, int total) {
  int idx = blockIdx.x * blockDim.x + threadIdx.x;
  if (idx < total) xb[idx] = (__bf16)x[idx];
}

__global__ void k_deg_init(float* __restrict__ deg, int n) {
  int i = blockIdx.x * blockDim.x + threadIdx.x;
  if (i < n) deg[i] = 1.0f;               // self-loop contribution
}

__global__ void k_deg_acc(const long long* __restrict__ ei, float* __restrict__ deg, int E) {
  int e = blockIdx.x * blockDim.x + threadIdx.x;
  if (e < E) unsafeAtomicAdd(&deg[ei[(size_t)E + e]], 1.0f);   // dst row
}

__global__ void k_dinv(float* __restrict__ deg, int n) {
  int i = blockIdx.x * blockDim.x + threadIdx.x;
  if (i < n) deg[i] = rsqrtf(deg[i]);     // deg >= 1 always
}

// ------------------------------ GCN GEMM -----------------------------------

// xw[N,128] = x_bf16[N,128] @ Wgcn_t  (one 16x16 tile per wave)
__global__ void k_gemm_xw(const __bf16* __restrict__ xb, const __bf16* __restrict__ wgt,
                          float* __restrict__ xw, int rowTiles) {
  const int lane = threadIdx.x & 31;
  const int w = blockIdx.x * (blockDim.x >> 5) + (threadIdx.x >> 5);
  if (w >= rowTiles * 8) return;          // wave-uniform guard
  const int rt = w >> 3, ct = w & 7;
  const __bf16* aBase = xb + (size_t)rt * 16 * FDIM;
  v8f acc = {};
  for (int kb = 0; kb < FDIM; kb += 32) {
    v16bf a = load_a_frag(aBase, FDIM, kb, lane);
    v16bf b = load_b_frag(wgt, FDIM, ct * 16, kb, lane);
    acc = __builtin_amdgcn_wmma_f32_16x16x32_bf16(false, a, false, b,
                                                  (short)0, acc, false, false);
  }
  const int half = lane >> 4, col = ct * 16 + (lane & 15);
  for (int k = 0; k < 8; ++k) {
    int row = rt * 16 + k + 8 * half;
    xw[(size_t)row * FDIM + col] = acc[k];
  }
}

// agg = dinv^2 * xw   (self-loop term; also fully initializes agg)
__global__ void k_self(const float* __restrict__ dinv, const float* __restrict__ xw,
                       float* __restrict__ agg, int total) {
  int idx = blockIdx.x * blockDim.x + threadIdx.x;
  if (idx >= total) return;
  float d = dinv[idx >> 7];
  agg[idx] = d * d * xw[idx];
}

// Edge scatter: one wave per edge, float4 per lane, native f32 atomics.
__global__ void k_scatter(const long long* __restrict__ ei, const float* __restrict__ dinv,
                          const float* __restrict__ xw, float* __restrict__ agg, int E) {
  const int lane = threadIdx.x & 31;
  const int e = blockIdx.x * (blockDim.x >> 5) + (threadIdx.x >> 5);
  if (e >= E) return;                     // wave-uniform guard
  const long long s = ei[e];
  const long long d = ei[(size_t)E + e];
  const float norm = dinv[s] * dinv[d];
  const float4 v = *(const float4*)(xw + (size_t)s * FDIM + lane * 4);
  float* dst = agg + (size_t)d * FDIM + lane * 4;
  unsafeAtomicAdd(dst + 0, v.x * norm);
  unsafeAtomicAdd(dst + 1, v.y * norm);
  unsafeAtomicAdd(dst + 2, v.z * norm);
  unsafeAtomicAdd(dst + 3, v.w * norm);
}

// zh[N,256] bf16 = [ relu(agg + b_gcn) , h_prev ]
__global__ void k_zh(const float* __restrict__ agg, const float* __restrict__ bg,
                     const float* __restrict__ h, __bf16* __restrict__ zh, int total) {
  int idx = blockIdx.x * blockDim.x + threadIdx.x;
  if (idx >= total) return;
  int node = idx >> 7, f = idx & 127;
  float z = agg[idx] + bg[f];
  z = z > 0.0f ? z : 0.0f;
  zh[(size_t)node * KCAT + f]        = (__bf16)z;
  zh[(size_t)node * KCAT + FDIM + f] = (__bf16)h[idx];
}

// ------------------------------ GRU GEMMs ----------------------------------

// r,u gates: two WMMA accumulators share A-fragments; emits u (f32) and r*h (bf16)
__global__ void k_ru(const __bf16* __restrict__ zh, const __bf16* __restrict__ wrt,
                     const __bf16* __restrict__ wut, const float* __restrict__ br,
                     const float* __restrict__ bu, const float* __restrict__ h,
                     __bf16* __restrict__ rh, float* __restrict__ ub, int rowTiles) {
  const int lane = threadIdx.x & 31;
  const int w = blockIdx.x * (blockDim.x >> 5) + (threadIdx.x >> 5);
  if (w >= rowTiles * 8) return;
  const int rt = w >> 3, ct = w & 7;
  const __bf16* aBase = zh + (size_t)rt * 16 * KCAT;
  v8f ar = {}, au = {};
  for (int kb = 0; kb < KCAT; kb += 32) {
    v16bf a  = load_a_frag(aBase, KCAT, kb, lane);
    v16bf b0 = load_b_frag(wrt, KCAT, ct * 16, kb, lane);
    ar = __builtin_amdgcn_wmma_f32_16x16x32_bf16(false, a, false, b0,
                                                 (short)0, ar, false, false);
    v16bf b1 = load_b_frag(wut, KCAT, ct * 16, kb, lane);
    au = __builtin_amdgcn_wmma_f32_16x16x32_bf16(false, a, false, b1,
                                                 (short)0, au, false, false);
  }
  const int half = lane >> 4, col = ct * 16 + (lane & 15);
  const float biasr = br[col], biasu = bu[col];
  for (int k = 0; k < 8; ++k) {
    int row = rt * 16 + k + 8 * half;
    size_t idx = (size_t)row * FDIM + col;
    float r = sigmoidf_(ar[k] + biasr);
    float u = sigmoidf_(au[k] + biasu);
    float hv = h[idx];
    rh[idx] = (__bf16)(r * hv);
    ub[idx] = u;
  }
}

// c gate + final blend: A = [z (from zh) | r*h (from rh)], h' = u*h + (1-u)*c
__global__ void k_ch(const __bf16* __restrict__ zh, const __bf16* __restrict__ rh,
                     const __bf16* __restrict__ wct, const float* __restrict__ bc,
                     const float* __restrict__ ub, const float* __restrict__ h,
                     float* __restrict__ out, int rowTiles) {
  const int lane = threadIdx.x & 31;
  const int w = blockIdx.x * (blockDim.x >> 5) + (threadIdx.x >> 5);
  if (w >= rowTiles * 8) return;
  const int rt = w >> 3, ct = w & 7;
  const __bf16* zBase = zh + (size_t)rt * 16 * KCAT;
  const __bf16* rBase = rh + (size_t)rt * 16 * FDIM;
  v8f ac = {};
  for (int ks = 0; ks < 8; ++ks) {
    const int kb = ks * 32;
    v16bf a = (ks < 4) ? load_a_frag(zBase, KCAT, kb, lane)
                       : load_a_frag(rBase, FDIM, kb - FDIM, lane);
    v16bf b = load_b_frag(wct, KCAT, ct * 16, kb, lane);
    ac = __builtin_amdgcn_wmma_f32_16x16x32_bf16(false, a, false, b,
                                                 (short)0, ac, false, false);
  }
  const int half = lane >> 4, col = ct * 16 + (lane & 15);
  const float biasc = bc[col];
  for (int k = 0; k < 8; ++k) {
    int row = rt * 16 + k + 8 * half;
    size_t idx = (size_t)row * FDIM + col;
    float c = tanhf(ac[k] + biasc);
    float u = ub[idx];
    float hv = h[idx];
    out[idx] = u * hv + (1.0f - u) * c;
  }
}

// ------------------------------- launcher ----------------------------------

extern "C" void kernel_launch(void* const* d_in, const int* in_sizes, int n_in,
                              void* d_out, int out_size, void* d_ws, size_t ws_size,
                              hipStream_t stream) {
  const float*     x_t    = (const float*)d_in[0];
  const float*     h_prev = (const float*)d_in[1];
  const long long* ei     = (const long long*)d_in[2];   // int64 [2,E]
  const float*     W_gcn  = (const float*)d_in[3];
  const float*     b_gcn  = (const float*)d_in[4];
  const float*     W_r    = (const float*)d_in[5];
  const float*     b_r    = (const float*)d_in[6];
  const float*     W_u    = (const float*)d_in[7];
  const float*     b_u    = (const float*)d_in[8];
  const float*     W_c    = (const float*)d_in[9];
  const float*     b_c    = (const float*)d_in[10];
  float*           out    = (float*)d_out;

  const int N = in_sizes[0] / FDIM;
  const int E = in_sizes[2] / 2;

  // workspace carve-out (all buffers fully written before use)
  char* ws = (char*)d_ws;
  size_t off = 0;
  auto carve = [&](size_t bytes) -> char* {
    off = (off + 255) & ~(size_t)255;
    char* p = ws + off;
    off += bytes;
    return p;
  };
  float*  deg  = (float*) carve((size_t)N * 4);
  float*  xw   = (float*) carve((size_t)N * FDIM * 4);
  float*  agg  = (float*) carve((size_t)N * FDIM * 4);
  float*  ubuf = (float*) carve((size_t)N * FDIM * 4);
  __bf16* xb   = (__bf16*)carve((size_t)N * FDIM * 2);
  __bf16* zh   = (__bf16*)carve((size_t)N * KCAT * 2);
  __bf16* rh   = (__bf16*)carve((size_t)N * FDIM * 2);
  __bf16* wgt  = (__bf16*)carve((size_t)FDIM * FDIM * 2);
  __bf16* wrt  = (__bf16*)carve((size_t)KCAT * FDIM * 2);
  __bf16* wut  = (__bf16*)carve((size_t)KCAT * FDIM * 2);
  __bf16* wct  = (__bf16*)carve((size_t)KCAT * FDIM * 2);
  (void)ws_size; (void)n_in; (void)out_size;

  const int T = 256;
  const int totalNF = N * FDIM;

  // weight conversion (small)
  k_cvt_weight<<<(FDIM * FDIM + T - 1) / T, T, 0, stream>>>(W_gcn, wgt, FDIM, FDIM);
  k_cvt_weight<<<(KCAT * FDIM + T - 1) / T, T, 0, stream>>>(W_r,   wrt, KCAT, FDIM);
  k_cvt_weight<<<(KCAT * FDIM + T - 1) / T, T, 0, stream>>>(W_u,   wut, KCAT, FDIM);
  k_cvt_weight<<<(KCAT * FDIM + T - 1) / T, T, 0, stream>>>(W_c,   wct, KCAT, FDIM);
  k_cast_x<<<(totalNF + T - 1) / T, T, 0, stream>>>(x_t, xb, totalNF);

  // degree / normalization
  k_deg_init<<<(N + T - 1) / T, T, 0, stream>>>(deg, N);
  k_deg_acc<<<(E + T - 1) / T, T, 0, stream>>>(ei, deg, E);
  k_dinv<<<(N + T - 1) / T, T, 0, stream>>>(deg, N);

  // GCN: GEMM -> self-loop init -> edge scatter -> z/h pack
  const int rowTiles = (N + 15) / 16;
  const int waves = rowTiles * 8;           // 8 column tiles of 16
  k_gemm_xw<<<(waves + 7) / 8, T, 0, stream>>>(xb, wgt, xw, rowTiles);
  k_self<<<(totalNF + T - 1) / T, T, 0, stream>>>(deg, xw, agg, totalNF);
  k_scatter<<<(E + 7) / 8, T, 0, stream>>>(ei, deg, xw, agg, E);
  k_zh<<<(totalNF + T - 1) / T, T, 0, stream>>>(agg, b_gcn, h_prev, zh, totalNF);

  // GRU: fused r/u then c/h
  k_ru<<<(waves + 7) / 8, T, 0, stream>>>(zh, wrt, wut, b_r, b_u, h_prev, rh, ubuf, rowTiles);
  k_ch<<<(waves + 7) / 8, T, 0, stream>>>(zh, rh, wct, b_c, ubuf, h_prev, out, rowTiles);
}
